// TransformerLayer_84516366450759
// MI455X (gfx1250) — matrix operand order repaired and checked by reading, not compiled, not run on hardware
//
#include <hip/hip_runtime.h>
#include <cstddef>
#include <cstdint>

// ---------------------------------------------------------------------------
// Transformer layer for MI455X (gfx1250): all GEMMs via V_WMMA_F32_16X16X32_F16
// ---------------------------------------------------------------------------

typedef __attribute__((ext_vector_type(16))) _Float16 v16h;
typedef __attribute__((ext_vector_type(8)))  _Float16 v8h;
typedef __attribute__((ext_vector_type(4)))  _Float16 v4h;
typedef __attribute__((ext_vector_type(8)))  float    v8f;
typedef __attribute__((ext_vector_type(4)))  float    v4f;

static constexpr int cB = 4;
static constexpr int cS = 2048;
static constexpr int cD = 1024;
static constexpr int cM = cB * cS;   // 8192 total rows

// epilogue selectors
static constexpr int EPI_F16            = 0; // store f16 row-major
static constexpr int EPI_F16T           = 1; // store f16 transposed (for V^T)
static constexpr int EPI_SCALE_F32      = 2; // store acc*scale as f32 (scores)
static constexpr int EPI_BIAS_RELU_F16  = 3; // relu(acc + bias) -> f16
static constexpr int EPI_RESID_F16      = 4; // acc + resid(f32) -> f16
static constexpr int EPI_BIAS_RESID_F32 = 5; // acc + bias + resid -> f32

// ---------------------------------------------------------------------------
// Generic batched WMMA GEMM: C = epilogue( A[M,K] * Bt[N,K]^T )
// Block = 256 threads = 8 waves (2 x 4); block tile 64M x 256N;
// wave tile 32M x 64N (2x4 accumulators); K step = 32.
// A is f16 row-major (or f32 row-major converted in-register when AF32).
// Bt is f16 with N rows of K contiguous elements (i.e. B stored transposed).
// ---------------------------------------------------------------------------
template <int EPI, bool AF32>
__global__ __launch_bounds__(256) void gemm_wmma_kernel(
    const void* __restrict__ Av, const _Float16* __restrict__ Btm,
    void* __restrict__ Cv,
    const float* __restrict__ bias, const float* __restrict__ resid,
    int K, int lda, int ldb, int ldc, int ldr,
    long long aBatch, long long bBatch, long long cBatch, long long rBatch,
    float scale)
{
  const int lane = threadIdx.x & 31;
  const int wave = threadIdx.x >> 5;
  const int wm   = wave >> 2;        // 0..1
  const int wn   = wave & 3;         // 0..3
  const int half = lane >> 4;        // 0/1 (lane-half)
  const int l16  = lane & 15;
  const int z    = blockIdx.z;

  const int tileM = blockIdx.y * 64  + wm * 32;
  const int tileN = blockIdx.x * 256 + wn * 64;

  const _Float16* Ah = AF32 ? nullptr : ((const _Float16*)Av + (size_t)z * aBatch);
  const float*    Af = AF32 ? ((const float*)Av + (size_t)z * aBatch) : nullptr;
  const _Float16* Bp = Btm + (size_t)z * bBatch;

  v8f acc[2][4];
#pragma unroll
  for (int mi = 0; mi < 2; ++mi)
#pragma unroll
    for (int nj = 0; nj < 4; ++nj)
      acc[mi][nj] = {};

  for (int kk = 0; kk < K; kk += 32) {
    // ---- A fragments: lane = M row; per-lane K = {kA..kA+7} U {kA+16..kA+23}
    const int kA = kk + half * 8;
    v16h afrag[2];
#pragma unroll
    for (int mi = 0; mi < 2; ++mi) {
      const int m = tileM + mi * 16 + l16;
      v16h a;
      if (AF32) {
        const float* ap = Af + (size_t)m * lda + kA;
        v4f f0 = *(const v4f*)(ap + 0);
        v4f f1 = *(const v4f*)(ap + 4);
        v4f f2 = *(const v4f*)(ap + 16);
        v4f f3 = *(const v4f*)(ap + 20);
#pragma unroll
        for (int e = 0; e < 4; ++e) {
          a[e]      = (_Float16)f0[e];
          a[4 + e]  = (_Float16)f1[e];
          a[8 + e]  = (_Float16)f2[e];
          a[12 + e] = (_Float16)f3[e];
        }
      } else {
        const _Float16* ap = Ah + (size_t)m * lda + kA;
        v8h lo = *(const v8h*)(ap);
        v8h hi = *(const v8h*)(ap + 16);
#pragma unroll
        for (int e = 0; e < 8; ++e) { a[e] = lo[e]; a[8 + e] = hi[e]; }
      }
      afrag[mi] = a;
    }
    // ---- B fragments: lane = N col; per-lane K = {kBB..kBB+7} U {kBB+16..kBB+23} in Bt
    const int kBB = kk + half * 8;
#pragma unroll
    for (int nj = 0; nj < 4; ++nj) {
      const int n = tileN + nj * 16 + l16;
      v16h b;
      {
        const _Float16* bp = Bp + (size_t)n * ldb + kBB;
        v8h blo = *(const v8h*)(bp);
        v8h bhi = *(const v8h*)(bp + 16);
#pragma unroll
        for (int e = 0; e < 8; ++e) { b[e] = blo[e]; b[8 + e] = bhi[e]; }
      }
#pragma unroll
      for (int mi = 0; mi < 2; ++mi) {
        acc[mi][nj] = __builtin_amdgcn_wmma_f32_16x16x32_f16(
            false, afrag[mi], false, b, (short)0, acc[mi][nj], false, false);
      }
    }
  }

  // ---- epilogue: C/D element r <-> (M = base + 8*half + r, N = l16)
#pragma unroll
  for (int mi = 0; mi < 2; ++mi) {
#pragma unroll
    for (int nj = 0; nj < 4; ++nj) {
      const int n     = tileN + nj * 16 + l16;
      const int mbase = tileM + mi * 16 + half * 8;
      v8f c = acc[mi][nj];
      if (EPI == EPI_F16) {
        _Float16* Ch = (_Float16*)Cv + (size_t)z * cBatch;
#pragma unroll
        for (int r = 0; r < 8; ++r)
          Ch[(size_t)(mbase + r) * ldc + n] = (_Float16)c[r];
      } else if (EPI == EPI_F16T) {
        _Float16* Ch = (_Float16*)Cv + (size_t)z * cBatch;
        v8h pk;
#pragma unroll
        for (int r = 0; r < 8; ++r) pk[r] = (_Float16)c[r];
        *(v8h*)(Ch + (size_t)n * ldc + mbase) = pk;   // one b128 store
      } else if (EPI == EPI_SCALE_F32) {
        float* Cf = (float*)Cv + (size_t)z * cBatch;
#pragma unroll
        for (int r = 0; r < 8; ++r)
          Cf[(size_t)(mbase + r) * ldc + n] = c[r] * scale;
      } else if (EPI == EPI_BIAS_RELU_F16) {
        _Float16* Ch = (_Float16*)Cv + (size_t)z * cBatch;
        const float bb = bias[n];
#pragma unroll
        for (int r = 0; r < 8; ++r) {
          float v = c[r] + bb;
          Ch[(size_t)(mbase + r) * ldc + n] = (_Float16)(v > 0.f ? v : 0.f);
        }
      } else if (EPI == EPI_RESID_F16) {
        _Float16* Ch = (_Float16*)Cv + (size_t)z * cBatch;
        const float* R = resid + (size_t)z * rBatch;
#pragma unroll
        for (int r = 0; r < 8; ++r)
          Ch[(size_t)(mbase + r) * ldc + n] =
              (_Float16)(c[r] + R[(size_t)(mbase + r) * ldr + n]);
      } else { // EPI_BIAS_RESID_F32
        float* Cf = (float*)Cv + (size_t)z * cBatch;
        const float* R = resid + (size_t)z * rBatch;
        const float bb = bias[n];
#pragma unroll
        for (int r = 0; r < 8; ++r)
          Cf[(size_t)(mbase + r) * ldc + n] =
              c[r] + bb + R[(size_t)(mbase + r) * ldr + n];
      }
    }
  }
}

// ---------------------------------------------------------------------------
// f32 -> f16 convert (vectorized x4)
// ---------------------------------------------------------------------------
__global__ __launch_bounds__(256) void cvt_f32_f16_kernel(
    const float* __restrict__ s, _Float16* __restrict__ d, size_t n)
{
  size_t i = ((size_t)blockIdx.x * blockDim.x + threadIdx.x) * 4;
  if (i + 3 < n) {
    v4f v = *(const v4f*)(s + i);
    v4h h;
#pragma unroll
    for (int e = 0; e < 4; ++e) h[e] = (_Float16)v[e];
    *(v4h*)(d + i) = h;
  }
}

// ---------------------------------------------------------------------------
// Transposed weight conversion: Wt[n][k] = (f16)W[k][n], LDS 32x32 tile
// ---------------------------------------------------------------------------
__global__ __launch_bounds__(256) void transpose_f16_kernel(
    const float* __restrict__ W, _Float16* __restrict__ Wt, int D)
{
  __shared__ _Float16 tile[32][33];
  const int bx = blockIdx.x * 32;   // n base
  const int by = blockIdx.y * 32;   // k base
  const int tx = threadIdx.x & 31;
  const int ty = threadIdx.x >> 5;  // 0..7
#pragma unroll
  for (int i = ty; i < 32; i += 8)
    tile[i][tx] = (_Float16)W[(size_t)(by + i) * D + (bx + tx)];
  __syncthreads();
#pragma unroll
  for (int i = ty; i < 32; i += 8)
    Wt[(size_t)(bx + i) * D + (by + tx)] = tile[tx][i];
}

// ---------------------------------------------------------------------------
// In-place row softmax over attn rows of length N (block = 256 threads / row)
// ---------------------------------------------------------------------------
__global__ __launch_bounds__(256) void softmax_rows_kernel(
    float* __restrict__ attn, int N)
{
  __shared__ float red[256];
  float* row = attn + (size_t)blockIdx.x * N;
  const int t = threadIdx.x;

  float mx = -3.4e38f;
  for (int i = t; i < N; i += 256) mx = fmaxf(mx, row[i]);
  red[t] = mx;
  __syncthreads();
  for (int s = 128; s > 0; s >>= 1) {
    if (t < s) red[t] = fmaxf(red[t], red[t + s]);
    __syncthreads();
  }
  mx = red[0];
  __syncthreads();

  float sum = 0.f;
  for (int i = t; i < N; i += 256) {
    float e = __expf(row[i] - mx);
    row[i] = e;
    sum += e;
  }
  red[t] = sum;
  __syncthreads();
  for (int s = 128; s > 0; s >>= 1) {
    if (t < s) red[t] += red[t + s];
    __syncthreads();
  }
  const float inv = 1.0f / red[0];
  for (int i = t; i < N; i += 256) row[i] *= inv;
}

// ---------------------------------------------------------------------------
extern "C" void kernel_launch(void* const* d_in, const int* in_sizes, int n_in,
                              void* d_out, int out_size, void* d_ws, size_t ws_size,
                              hipStream_t stream) {
  (void)in_sizes; (void)n_in; (void)out_size; (void)ws_size;

  const float* X  = (const float*)d_in[0];
  const float* Wq = (const float*)d_in[1];
  const float* Wk = (const float*)d_in[2];
  const float* Wv = (const float*)d_in[3];
  const float* W1 = (const float*)d_in[4];
  const float* b1 = (const float*)d_in[5];
  const float* W2 = (const float*)d_in[6];
  const float* b2 = (const float*)d_in[7];

  float* out  = (float*)d_out;                 // [B*S, D] f32
  float* attn = out + (size_t)cM * cD;         // [B, S, S] f32 (second output)

  // workspace carve-up (f16, ~106 MiB total)
  char* p = (char*)d_ws;
  auto take = [&](size_t elems) { _Float16* r = (_Float16*)p; p += elems * 2; return r; };
  _Float16* Xh  = take((size_t)cM * cD);
  _Float16* Wqt = take((size_t)cD * cD);
  _Float16* Wkt = take((size_t)cD * cD);
  _Float16* Wvt = take((size_t)cD * cD);
  _Float16* W1t = take((size_t)cD * cD);
  _Float16* W2t = take((size_t)cD * cD);
  _Float16* Qh  = take((size_t)cM * cD);
  _Float16* Kh  = take((size_t)cM * cD);
  _Float16* Vt  = take((size_t)cD * cM);       // V^T: [D][B*S]
  _Float16* R1h = take((size_t)cM * cD);       // f16(attn_out + x)
  _Float16* Hh  = take((size_t)cM * cD);       // f16(relu(h@W1 + b1))

  // 1) conversions
  {
    size_t n = (size_t)cM * cD;
    cvt_f32_f16_kernel<<<dim3((unsigned)(n / 4 / 256)), 256, 0, stream>>>(X, Xh, n);
  }
  {
    dim3 tg(cD / 32, cD / 32);
    transpose_f16_kernel<<<tg, 256, 0, stream>>>(Wq, Wqt, cD);
    transpose_f16_kernel<<<tg, 256, 0, stream>>>(Wk, Wkt, cD);
    transpose_f16_kernel<<<tg, 256, 0, stream>>>(Wv, Wvt, cD);
    transpose_f16_kernel<<<tg, 256, 0, stream>>>(W1, W1t, cD);
    transpose_f16_kernel<<<tg, 256, 0, stream>>>(W2, W2t, cD);
  }

  // 2) projections: Q,K row-major f16; V stored transposed
  {
    dim3 g(cD / 256, cM / 64, 1);
    gemm_wmma_kernel<EPI_F16, false><<<g, 256, 0, stream>>>(
        Xh, Wqt, Qh, nullptr, nullptr, cD, cD, cD, cD, 0, 0, 0, 0, 0, 1.0f);
    gemm_wmma_kernel<EPI_F16, false><<<g, 256, 0, stream>>>(
        Xh, Wkt, Kh, nullptr, nullptr, cD, cD, cD, cD, 0, 0, 0, 0, 0, 1.0f);
    gemm_wmma_kernel<EPI_F16T, false><<<g, 256, 0, stream>>>(
        Xh, Wvt, Vt, nullptr, nullptr, cD, cD, cD, cM /*ldc=B*S*/, 0, 0, 0, 0, 0, 1.0f);
  }

  // 3) scores = (Q . K^T) / 32 -> attn (f32), batched over B
  {
    dim3 g(cS / 256, cS / 64, cB);
    gemm_wmma_kernel<EPI_SCALE_F32, false><<<g, 256, 0, stream>>>(
        Qh, Kh, attn, nullptr, nullptr, cD, cD, cD, cS, 0,
        (long long)cS * cD, (long long)cS * cD, (long long)cS * cS, 0, 0.03125f);
  }

  // 4) softmax rows in place (attn is a final output)
  softmax_rows_kernel<<<dim3(cM), 256, 0, stream>>>(attn, cS);

  // 5) attn_out + x -> R1h (f16); A = attn (f32, converted in-register)
  {
    dim3 g(cD / 256, cS / 64, cB);
    gemm_wmma_kernel<EPI_RESID_F16, true><<<g, 256, 0, stream>>>(
        attn, Vt, R1h, nullptr, X, cS, cS /*lda*/, cM /*ldb*/, cD /*ldc*/, cD /*ldr*/,
        (long long)cS * cS, (long long)cS /*Vt col offset*/,
        (long long)cS * cD, (long long)cS * cD, 1.0f);
  }

  // 6) h = relu(R1 @ W1 + b1) -> Hh (f16)
  {
    dim3 g(cD / 256, cM / 64, 1);
    gemm_wmma_kernel<EPI_BIAS_RELU_F16, false><<<g, 256, 0, stream>>>(
        R1h, W1t, Hh, b1, nullptr, cD, cD, cD, cD, 0, 0, 0, 0, 0, 1.0f);
  }

  // 7) out = Hh @ W2 + b2 + x (f32)
  {
    dim3 g(cD / 256, cM / 64, 1);
    gemm_wmma_kernel<EPI_BIAS_RESID_F32, false><<<g, 256, 0, stream>>>(
        Hh, W2t, out, b2, X, cD, cD, cD, cD, cD, 0, 0, 0, 0, 1.0f);
  }
}
